// MultiHeadAttention_48017734369564
// MI455X (gfx1250) — matrix operand order repaired and checked by reading, not compiled, not run on hardware
//
#include <hip/hip_runtime.h>
#include <hip/hip_bf16.h>

// ---------------------------------------------------------------- constants
constexpr int Bc = 4, Tc = 2048, Dc = 512, Hc = 8, DKc = 64;
constexpr int Mc = Bc * Tc;          // 8192 rows for all projections

// ---------------------------------------------------------------- types
typedef __attribute__((ext_vector_type(16))) __bf16 bf16x16;
typedef __attribute__((ext_vector_type(8)))  float  f32x8;
typedef __attribute__((ext_vector_type(4)))  unsigned int u32x4;
typedef __attribute__((ext_vector_type(8)))  unsigned int u32x8;

union FragBF {
    bf16x16 v;
    uint4   u4[2];
    unsigned short s[16];
};

__device__ __forceinline__ unsigned short f2bf(float f) {
    union { float f; unsigned u; } x; x.f = f;
    unsigned u = x.u;
    unsigned r = u + 0x7FFFu + ((u >> 16) & 1u);   // round-to-nearest-even
    return (unsigned short)(r >> 16);
}

// ---------------------------------------------------------------- fp32 -> bf16
__global__ __launch_bounds__(256)
void cvt_f32_bf16(const float* __restrict__ in, unsigned short* __restrict__ out, int n) {
    int i = (blockIdx.x * blockDim.x + threadIdx.x) * 4;
    if (i < n) {
        float4 f = *(const float4*)(in + i);
        unsigned lo = (unsigned)f2bf(f.x) | ((unsigned)f2bf(f.y) << 16);
        unsigned hi = (unsigned)f2bf(f.z) | ((unsigned)f2bf(f.w) << 16);
        uint2 p; p.x = lo; p.y = hi;
        *(uint2*)(out + i) = p;
    }
}

// ---------------------------------------------------------------- WMMA GEMM
// Out[m,n] = sum_k A[m,k] * W[n,k] + bias[n]      (torch Linear: x @ W.T + b)
// mode 0 : bf16 output scattered to [B,H,T,DK]   (QKV projections)
// mode 1 : fp32 output row-major [M,N]           (final projection -> d_out)
#define TILE_M 128
#define TILE_N 128
#define TILE_K 32
#define LDSA   40      // 32 + 8 pad (bf16 elems), keeps 16B alignment

__global__ __launch_bounds__(256)
void gemm_bf16_wmma(const unsigned short* __restrict__ A,   // [M][512] bf16
                    const unsigned short* __restrict__ W,   // [512][512] bf16 (n-major)
                    const float* __restrict__ bias,         // [512] fp32
                    void* __restrict__ Out, int mode)
{
    __shared__ __align__(16) unsigned short As[TILE_M * LDSA];
    __shared__ __align__(16) unsigned short Bs[TILE_N * LDSA];

    const int tid  = threadIdx.x;
    const int lane = tid & 31;
    const int wave = tid >> 5;
    const int half = lane >> 4;
    const int ln   = lane & 15;

    const int bm = blockIdx.y * TILE_M;
    const int bn = blockIdx.x * TILE_N;
    const int wm = (wave >> 1) * 32;     // 4 waves along M
    const int wn = (wave & 1) * 64;      // 2 waves along N

    f32x8 acc[2][4];
#pragma unroll
    for (int i = 0; i < 2; ++i)
#pragma unroll
        for (int j = 0; j < 4; ++j)
            acc[i][j] = (f32x8){0.f,0.f,0.f,0.f,0.f,0.f,0.f,0.f};

    const int lrow = tid >> 1;           // 0..127
    const int lseg = (tid & 1) * 16;     // bf16 elem offset 0 / 16

    for (int k0 = 0; k0 < Dc; k0 += TILE_K) {
        // ---- cooperative stage: 32B per thread per tile
        const uint4* gA = (const uint4*)(A + (size_t)(bm + lrow) * Dc + k0 + lseg);
        uint4 a0 = gA[0], a1 = gA[1];
        const uint4* gB = (const uint4*)(W + (size_t)(bn + lrow) * Dc + k0 + lseg);
        uint4 b0 = gB[0], b1 = gB[1];
        *(uint4*)(&As[lrow * LDSA + lseg])     = a0;
        *(uint4*)(&As[lrow * LDSA + lseg + 8]) = a1;
        *(uint4*)(&Bs[lrow * LDSA + lseg])     = b0;
        *(uint4*)(&Bs[lrow * LDSA + lseg + 8]) = b1;
        if (k0 + TILE_K < Dc) {   // prefetch next K slice -> global_prefetch_b8
            __builtin_prefetch(A + (size_t)(bm + lrow) * Dc + k0 + TILE_K + lseg, 0, 3);
            __builtin_prefetch(W + (size_t)(bn + lrow) * Dc + k0 + TILE_K + lseg, 0, 3);
        }
        __syncthreads();

        // ---- fragments (ISA 7.12.2 layouts)
        FragBF af[2], bf[4];
#pragma unroll
        for (int mt = 0; mt < 2; ++mt) {
            const unsigned short* p = &As[(wm + mt * 16 + ln) * LDSA + half * 8];
            af[mt].u4[0] = *(const uint4*)(p);        // K = half*8 .. +7
            af[mt].u4[1] = *(const uint4*)(p + 16);   // K = half*8+16 ..
        }
#pragma unroll
        for (int nt = 0; nt < 4; ++nt) {
            const unsigned short* p = &Bs[(wn + nt * 16 + ln) * LDSA + half * 8];
            bf[nt].u4[0] = *(const uint4*)(p);        // K = half*8 .. +7
            bf[nt].u4[1] = *(const uint4*)(p + 16);   // K = half*8+16 ..
        }
#pragma unroll
        for (int mt = 0; mt < 2; ++mt)
#pragma unroll
            for (int nt = 0; nt < 4; ++nt)
                acc[mt][nt] = __builtin_amdgcn_wmma_f32_16x16x32_bf16(
                    false, af[mt].v, false, bf[nt].v, (short)0, acc[mt][nt],
                    false, false);
        __syncthreads();
    }

    // ---- epilogue
#pragma unroll
    for (int mt = 0; mt < 2; ++mt) {
#pragma unroll
        for (int nt = 0; nt < 4; ++nt) {
            const int col = bn + wn + nt * 16 + ln;
            const float bv = bias[col];
#pragma unroll
            for (int r = 0; r < 8; ++r) {
                const int row = bm + wm + mt * 16 + half * 8 + r;
                float v = acc[mt][nt][r] + bv;
                if (mode == 0) {  // bf16, scatter to [B,H,T,DK]
                    int b = row >> 11, t = row & (Tc - 1);
                    int h = col >> 6,  dk = col & (DKc - 1);
                    ((unsigned short*)Out)[(((size_t)(b * Hc + h)) * Tc + t) * DKc + dk] = f2bf(v);
                } else {          // fp32 row-major -> d_out
                    ((float*)Out)[(size_t)row * Dc + col] = v;
                }
            }
        }
    }
}

// ---------------------------------------------------------------- flash attention
// grid = (B*H, T/128); block = 256 (8 waves); each wave: 16 query rows x DK=64.
#define KVS 72   // 64 + 8 pad (bf16 elems) == TDM pad: 32 dwords + 4 dwords
#define VTS 40   // 32 + 8 pad
#define PSS 40   // 32 + 8 pad

__global__ __launch_bounds__(256)
void flash_attn_wmma(const unsigned short* __restrict__ Q,   // [B,H,T,DK] bf16
                     const unsigned short* __restrict__ Km,
                     const unsigned short* __restrict__ V,
                     unsigned short* __restrict__ Ctx)       // [B,T,D] bf16
{
    __shared__ __align__(16) unsigned short Ks[32 * KVS];        // [key][feat] (TDM-filled)
    __shared__ __align__(16) unsigned short Vt[64 * VTS];        // [feat][key] (transposed)
    __shared__ __align__(16) unsigned short Ps[8 * 16 * PSS];    // per-wave P staging

    const int tid  = threadIdx.x;
    const int lane = tid & 31;
    const int wave = tid >> 5;
    const int half = lane >> 4;
    const int ln   = lane & 15;

    const int bh = blockIdx.x;                       // b*H + h
    const size_t headBase = (size_t)bh * Tc * DKc;
    const int qbase = blockIdx.y * 128 + wave * 16;

    // ---- Q A-fragments (held for whole kernel), 2 K-steps over DK=64
    FragBF qf[2];
    const unsigned short* qp = Q + headBase + (size_t)(qbase + ln) * DKc;
#pragma unroll
    for (int s = 0; s < 2; ++s) {
        qf[s].u4[0] = *(const uint4*)(qp + s * 32 + half * 8);
        qf[s].u4[1] = *(const uint4*)(qp + s * 32 + half * 8 + 16);
    }

    f32x8 acc[4];
#pragma unroll
    for (int nt = 0; nt < 4; ++nt) acc[nt] = (f32x8){0.f,0.f,0.f,0.f,0.f,0.f,0.f,0.f};
    float mrow[8], lsum[8];
#pragma unroll
    for (int r = 0; r < 8; ++r) { mrow[r] = -1e30f; lsum[r] = 0.f; }
    const float scale = 0.125f;  // 1/sqrt(64)

    for (int kt = 0; kt < Tc; kt += 32) {
        // ---- K tile [32 keys][64 feats] via Tensor Data Mover (async DMA -> LDS)
        // D# group0: count=1 | lds_addr | global_addr | type=2
        // D# group1: data_size=2B, pad_enable (32 dwords + 4 dwords -> stride 72 bf16),
        //            tensor = [T x DK], tile = [32 x 64], dim0_stride = DK.
        if (wave == 0) {
            const unsigned long long gaddr =
                (unsigned long long)(uintptr_t)(Km + headBase + (size_t)kt * DKc);
            const unsigned ldsoff = (unsigned)(uintptr_t)(&Ks[0]);
            u32x4 g0;
            g0[0] = 1u;                                        // count=1 (valid user D#)
            g0[1] = ldsoff;                                    // lds_addr (bytes)
            g0[2] = (unsigned)(gaddr & 0xFFFFFFFFu);           // global_addr[31:0]
            g0[3] = (unsigned)((gaddr >> 32) & 0x01FFFFFFu)    // global_addr[56:32]
                  | 0x80000000u;                               // type=2 ("image")
            u32x8 g1;
            g1[0] = (1u << 16)        // data_size: 1 -> 2 bytes
                  | (1u << 20)        // pad_enable
                  | (4u << 22)        // pad_interval: 2^(4+1)=32 dwords (one 64-bf16 row)
                  | (3u << 25);       // pad_amount: 4 dwords (8 bf16)
            g1[1] = (unsigned)DKc << 16;     // tensor_dim0[15:0]=64 (bits 63:48)
            g1[2] = (unsigned)Tc  << 16;     // dim0 hi=0 | tensor_dim1[15:0]=2048
            g1[3] = (unsigned)DKc << 16;     // dim1 hi=0 | tile_dim0=64
            g1[4] = 32u;                     // tile_dim1=32 keys | tile_dim2=0
            g1[5] = (unsigned)DKc;           // tensor_dim0_stride[31:0]=64
            g1[6] = 0u;                      // stride hi | tensor_dim1_stride lo
            g1[7] = 0u;
            asm volatile("tensor_load_to_lds %0, %1" :: "s"(g0), "s"(g1) : "memory");
            __builtin_amdgcn_s_wait_tensorcnt(0);
        }
        // ---- V tile transposed [64 feats][32 keys], cooperative stage
        {
            const int key = tid >> 3;
            const int seg = (tid & 7) * 8;
            uint4 vv = *(const uint4*)(V + headBase + (size_t)(kt + key) * DKc + seg);
            const unsigned short* ve = (const unsigned short*)&vv;
#pragma unroll
            for (int j = 0; j < 8; ++j)
                Vt[(seg + j) * VTS + key] = ve[j];
        }
        __syncthreads();

        // ---- S = (Q K^T) * scale : 16 rows x 32 keys  (4 WMMAs)
        f32x8 s[2];
        s[0] = (f32x8){0.f,0.f,0.f,0.f,0.f,0.f,0.f,0.f};
        s[1] = s[0];
#pragma unroll
        for (int nt = 0; nt < 2; ++nt) {
#pragma unroll
            for (int ks = 0; ks < 2; ++ks) {
                FragBF kf;
                const unsigned short* p = &Ks[(nt * 16 + ln) * KVS + ks * 32 + half * 8];
                kf.u4[0] = *(const uint4*)(p);
                kf.u4[1] = *(const uint4*)(p + 16);
                s[nt] = __builtin_amdgcn_wmma_f32_16x16x32_bf16(
                    false, qf[ks].v, false, kf.v, (short)0, s[nt], false, false);
            }
        }

        // ---- online softmax (fp32 VALU, co-executes with XDL)
#pragma unroll
        for (int r = 0; r < 8; ++r) {
            float s0 = s[0][r] * scale, s1 = s[1][r] * scale;
            float mx = fmaxf(s0, s1);
#pragma unroll
            for (int off = 8; off >= 1; off >>= 1)
                mx = fmaxf(mx, __shfl_xor(mx, off, 16));
            float mnew = fmaxf(mrow[r], mx);
            float p0 = __expf(s0 - mnew);
            float p1 = __expf(s1 - mnew);
            float rs = p0 + p1;
#pragma unroll
            for (int off = 8; off >= 1; off >>= 1)
                rs += __shfl_xor(rs, off, 16);
            float alpha = __expf(mrow[r] - mnew);
            lsum[r] = lsum[r] * alpha + rs;
            mrow[r] = mnew;
            s[0][r] = p0; s[1][r] = p1;
#pragma unroll
            for (int nt = 0; nt < 4; ++nt) acc[nt][r] *= alpha;
        }

        // ---- re-shape P (C-layout) -> A-fragment via per-wave LDS
        unsigned short* pw = &Ps[wave * 16 * PSS];
#pragma unroll
        for (int r = 0; r < 8; ++r) {
            const int row = half * 8 + r;
            pw[row * PSS + ln]      = f2bf(s[0][r]);
            pw[row * PSS + 16 + ln] = f2bf(s[1][r]);
        }
        __builtin_amdgcn_wave_barrier();
        asm volatile("s_wait_dscnt 0" ::: "memory");   // LDS RAW inside wave

        FragBF pf;
        {
            const unsigned short* p = &pw[ln * PSS + half * 8];
            pf.u4[0] = *(const uint4*)(p);
            pf.u4[1] = *(const uint4*)(p + 16);
        }

        // ---- acc += P @ V   (k=32 keys, 4 feature tiles)
#pragma unroll
        for (int nt = 0; nt < 4; ++nt) {
            FragBF vf;
            const unsigned short* p = &Vt[(nt * 16 + ln) * VTS + half * 8];
            vf.u4[0] = *(const uint4*)(p);
            vf.u4[1] = *(const uint4*)(p + 16);
            acc[nt] = __builtin_amdgcn_wmma_f32_16x16x32_bf16(
                false, pf.v, false, vf.v, (short)0, acc[nt], false, false);
        }
        __syncthreads();
    }

    // ---- normalize + store ctx as bf16 [B,T,D]
    const int b = bh >> 3, h = bh & 7;
#pragma unroll
    for (int nt = 0; nt < 4; ++nt) {
#pragma unroll
        for (int r = 0; r < 8; ++r) {
            const int row = qbase + half * 8 + r;
            float v = acc[nt][r] / lsum[r];
            Ctx[((size_t)(b * Tc + row)) * Dc + h * DKc + nt * 16 + ln] = f2bf(v);
        }
    }
}

// ---------------------------------------------------------------- host launch
extern "C" void kernel_launch(void* const* d_in, const int* in_sizes, int n_in,
                              void* d_out, int out_size, void* d_ws, size_t ws_size,
                              hipStream_t stream) {
    const float* x  = (const float*)d_in[0];
    const float* Wq = (const float*)d_in[1];
    const float* bq = (const float*)d_in[2];
    const float* Wk = (const float*)d_in[3];
    const float* bk = (const float*)d_in[4];
    const float* Wv = (const float*)d_in[5];
    const float* bv = (const float*)d_in[6];
    const float* Wo = (const float*)d_in[7];
    const float* bo = (const float*)d_in[8];

    char* p = (char*)d_ws;
    auto take = [&](size_t bytes) { char* r = p; p += (bytes + 255) & ~(size_t)255; return r; };
    unsigned short* xb  = (unsigned short*)take((size_t)Mc * Dc * 2);
    unsigned short* wqb = (unsigned short*)take((size_t)Dc * Dc * 2);
    unsigned short* wkb = (unsigned short*)take((size_t)Dc * Dc * 2);
    unsigned short* wvb = (unsigned short*)take((size_t)Dc * Dc * 2);
    unsigned short* wob = (unsigned short*)take((size_t)Dc * Dc * 2);
    unsigned short* Qb  = (unsigned short*)take((size_t)Mc * Dc * 2);
    unsigned short* Kb  = (unsigned short*)take((size_t)Mc * Dc * 2);
    unsigned short* Vb  = (unsigned short*)take((size_t)Mc * Dc * 2);
    unsigned short* Cx  = (unsigned short*)take((size_t)Mc * Dc * 2);

    // 1) fp32 -> bf16
    cvt_f32_bf16<<<(Mc * Dc) / 4 / 256, 256, 0, stream>>>(x, xb, Mc * Dc);
    cvt_f32_bf16<<<(Dc * Dc) / 4 / 256, 256, 0, stream>>>(Wq, wqb, Dc * Dc);
    cvt_f32_bf16<<<(Dc * Dc) / 4 / 256, 256, 0, stream>>>(Wk, wkb, Dc * Dc);
    cvt_f32_bf16<<<(Dc * Dc) / 4 / 256, 256, 0, stream>>>(Wv, wvb, Dc * Dc);
    cvt_f32_bf16<<<(Dc * Dc) / 4 / 256, 256, 0, stream>>>(Wo, wob, Dc * Dc);

    // 2) QKV projections (WMMA GEMM, bf16 epilogue -> [B,H,T,DK])
    dim3 gGrid(Dc / TILE_N, Mc / TILE_M);   // (4, 64)
    gemm_bf16_wmma<<<gGrid, 256, 0, stream>>>(xb, wqb, bq, Qb, 0);
    gemm_bf16_wmma<<<gGrid, 256, 0, stream>>>(xb, wkb, bk, Kb, 0);
    gemm_bf16_wmma<<<gGrid, 256, 0, stream>>>(xb, wvb, bv, Vb, 0);

    // 3) fused flash attention (TDM-staged K tiles)
    dim3 fGrid(Bc * Hc, Tc / 128);          // (32, 16)
    flash_attn_wmma<<<fGrid, 256, 0, stream>>>(Qb, Kb, Vb, Cx);

    // 4) output projection, fp32 epilogue straight into d_out
    gemm_bf16_wmma<<<gGrid, 256, 0, stream>>>(Cx, wob, bo, d_out, 1);
}
